// InternalInteraction_9079560863790
// MI455X (gfx1250) — matrix operand order repaired and checked by reading, not compiled, not run on hardware
//
#include <hip/hip_runtime.h>

// B=2048, A=16, D=128, H=512
#define A_DIM 16
#define D_DIM 128
#define H_DIM 512

#define XPITCH 136    // bf16/row of xs   (272B rows, 16B aligned)
#define HPITCH 520    // bf16/row of hsh  (1040B rows, 16B aligned)
#define OPITCH 132    // f32/row of outbuf

#define NPAIR 136     // unique unordered pairs (16*17/2)
#define CH 3          // M-tiles (of 16 pair-rows) per chunk
#define NCHUNK 3      // 3*3*16 = 144 pair-rows >= 136
#define CROWS (CH * 16)

typedef __attribute__((ext_vector_type(16))) __bf16 v16bf;
typedef __attribute__((ext_vector_type(8)))  __bf16 v8bf;
typedef __attribute__((ext_vector_type(8)))  float  v8f;

__device__ __forceinline__ __bf16 f2bf(float f) { return (__bf16)f; }

__device__ __forceinline__ v16bf cat8(v8bf lo, v8bf hi) {
    return __builtin_shufflevector(lo, hi, 0, 1, 2, 3, 4, 5, 6, 7,
                                           8, 9, 10, 11, 12, 13, 14, 15);
}

__global__ __launch_bounds__(256, 1)
void interaction_mlp_kernel(const float* __restrict__ x,   // [B,16,128]
                            const float* __restrict__ W1,  // [512,128]
                            const float* __restrict__ b1,  // [512]
                            const float* __restrict__ W2,  // [128,512]
                            const float* __restrict__ b2,  // [128]
                            float* __restrict__ out)       // [B,16,128]
{
    __shared__ __bf16 xs[A_DIM * XPITCH];         // x[b] as bf16
    __shared__ __bf16 hsh[2][CROWS * HPITCH];     // h^T chunk, [pair][h]
    __shared__ float  outbuf[A_DIM * OPITCH];     // accumulated out[j][d]

    const int b    = blockIdx.x;
    const int tid  = threadIdx.x;
    const int w    = tid >> 5;    // wave id 0..7
    const int lane = tid & 31;
    const int nl   = lane & 15;
    const int half = lane >> 4;

    // ---- stage x[b] into LDS as bf16; zero outbuf ----
    {
        const float* xb = x + (size_t)b * (A_DIM * D_DIM);
        const int e0  = tid * 8;
        const int row = e0 >> 7;
        const int col = e0 & 127;
        float4 f0 = *(const float4*)(xb + e0);
        float4 f1 = *(const float4*)(xb + e0 + 4);
        __bf16* dst = &xs[row * XPITCH + col];
        dst[0] = f2bf(f0.x); dst[1] = f2bf(f0.y);
        dst[2] = f2bf(f0.z); dst[3] = f2bf(f0.w);
        dst[4] = f2bf(f1.x); dst[5] = f2bf(f1.y);
        dst[6] = f2bf(f1.z); dst[7] = f2bf(f1.w);
    }
    for (int k = tid; k < A_DIM * OPITCH; k += 256) outbuf[k] = 0.0f;

    // ---- persistent weight fragments in registers (A-operand layout) ----
    // A layout (16x32 bf16): lane row m = lane&15,
//   element e -> K = 16*half + e (this code's k order; B uses the same).
    v16bf w1f[4][4];                      // GEMM1: A[m=h][k=d] = W1[h][d]
#pragma unroll
    for (int kt = 0; kt < 4; ++kt) {
#pragma unroll
        for (int ntl = 0; ntl < 4; ++ntl) {
            const int h = (4 * w + ntl) * 16 + nl;
            const float* src = W1 + (size_t)h * D_DIM + kt * 32;
#pragma unroll
            for (int e = 0; e < 8; ++e) {
                w1f[kt][ntl][e]     = f2bf(src[16 * half + e]);
                w1f[kt][ntl][e + 8] = f2bf(src[16 * half + 8 + e]);
            }
        }
    }
    v16bf w2f[16];                        // GEMM2: A[m=d][k=h] = W2[d][h]
#pragma unroll
    for (int kc = 0; kc < 16; ++kc) {
        const int d = w * 16 + nl;
        const float* src = W2 + (size_t)d * H_DIM + kc * 32;
#pragma unroll
        for (int e = 0; e < 8; ++e) {
            w2f[kc][e]     = f2bf(src[16 * half + e]);
            w2f[kc][e + 8] = f2bf(src[16 * half + 8 + e]);
        }
    }

    // bias rows match C-tile rows (m = r + 8*half)
    v8f b1v[4];
#pragma unroll
    for (int ntl = 0; ntl < 4; ++ntl) {
        const float* p = b1 + (4 * w + ntl) * 16 + 8 * half;
#pragma unroll
        for (int r = 0; r < 8; ++r) b1v[ntl][r] = p[r];
    }

    __syncthreads();

#pragma unroll 1
    for (int ch = 0; ch < NCHUNK; ++ch) {
        __bf16* hbuf = hsh[ch & 1];

        // ---- decode this chunk's pair (i,j) per lane (p = i*16 - i(i-1)/2 + j-i) ----
        int ip[CH], jp[CH];
        bool valid[CH];
#pragma unroll
        for (int m = 0; m < CH; ++m) {
            const int p = (ch * CH + m) * 16 + nl;
            const bool v = (p < NPAIR);
            int pp = v ? p : 0;
            int ii = 0;
            while (pp >= 16 - ii) { pp -= 16 - ii; ++ii; }
            ip[m] = ii; jp[m] = ii + pp; valid[m] = v;
        }

        // ---- GEMM1: h^T tiles [16h x 16pair] = W1 * P^T, bias in C, relu ----
#pragma unroll
        for (int m = 0; m < CH; ++m) {
            // P^T fragment (B layout): lane col = pair nl, K(d) = 16*half + e
            v16bf a1[4];
#pragma unroll
            for (int kt = 0; kt < 4; ++kt) {
                const v8bf* xi = (const v8bf*)(xs + ip[m] * XPITCH + kt * 32 + 16 * half);
                const v8bf* xj = (const v8bf*)(xs + jp[m] * XPITCH + kt * 32 + 16 * half);
                a1[kt] = cat8(xi[0] * xj[0], xi[1] * xj[1]);   // v_pk_mul_bf16
            }
#pragma unroll
            for (int ntl = 0; ntl < 4; ++ntl) {
                v8f c = b1v[ntl];
#pragma unroll
                for (int kt = 0; kt < 4; ++kt)
                    c = __builtin_amdgcn_wmma_f32_16x16x32_bf16(
                            false, w1f[kt][ntl], false, a1[kt], (short)0, c,
                            false, false);
                v8bf hv;
#pragma unroll
                for (int r = 0; r < 8; ++r) {
                    float v = c[r];
                    hv[r] = f2bf(v > 0.0f ? v : 0.0f);
                }
                *(v8bf*)(hbuf + (m * 16 + nl) * HPITCH +
                         (4 * w + ntl) * 16 + 8 * half) = hv;
            }
        }
        __syncthreads();

        // ---- GEMM2: acc^T [16d x 16pair] = W2 * h^T; scatter-add to outbuf ----
#pragma unroll
        for (int m = 0; m < CH; ++m) {
            v8f acc = {};
#pragma unroll
            for (int kc = 0; kc < 16; ++kc) {
                const v8bf* hp = (const v8bf*)(hbuf + (m * 16 + nl) * HPITCH +
                                               kc * 32 + 16 * half);
                v16bf a2 = cat8(hp[0], hp[1]);
                acc = __builtin_amdgcn_wmma_f32_16x16x32_bf16(
                          false, w2f[kc], false, a2, (short)0, acc, false, false);
            }
            // lane: pair col = nl, d = w*16 + 8*half + r (wave-exclusive columns)
            if (valid[m]) {
                float* oi = outbuf + ip[m] * OPITCH + w * 16 + 8 * half;
#pragma unroll
                for (int r = 0; r < 8; ++r) atomicAdd(&oi[r], acc[r]);  // ds_add_f32
                if (jp[m] != ip[m]) {
                    float* oj = outbuf + jp[m] * OPITCH + w * 16 + 8 * half;
#pragma unroll
                    for (int r = 0; r < 8; ++r) atomicAdd(&oj[r], acc[r]);
                }
            }
        }
        // next chunk writes the other h buffer: one barrier per chunk suffices
    }

    __syncthreads();

    // ---- out[b][j][d] = outbuf[j][d] + 16*b2[d] ----
    {
        const int j  = tid >> 4;
        const int d0 = (tid & 15) * 8;
        const float* ob = outbuf + j * OPITCH + d0;
        float* og = out + (size_t)b * (A_DIM * D_DIM) + j * D_DIM + d0;
        float4 r0, r1;
        r0.x = ob[0] + 16.0f * b2[d0 + 0]; r0.y = ob[1] + 16.0f * b2[d0 + 1];
        r0.z = ob[2] + 16.0f * b2[d0 + 2]; r0.w = ob[3] + 16.0f * b2[d0 + 3];
        r1.x = ob[4] + 16.0f * b2[d0 + 4]; r1.y = ob[5] + 16.0f * b2[d0 + 5];
        r1.z = ob[6] + 16.0f * b2[d0 + 6]; r1.w = ob[7] + 16.0f * b2[d0 + 7];
        *(float4*)(og)     = r0;
        *(float4*)(og + 4) = r1;
    }
}

extern "C" void kernel_launch(void* const* d_in, const int* in_sizes, int n_in,
                              void* d_out, int out_size, void* d_ws, size_t ws_size,
                              hipStream_t stream) {
    const float* x  = (const float*)d_in[0];   // [2048,16,128]
    const float* W1 = (const float*)d_in[1];   // [512,128]
    const float* b1 = (const float*)d_in[2];   // [512]
    const float* W2 = (const float*)d_in[3];   // [128,512]
    const float* b2 = (const float*)d_in[4];   // [128]
    float* out = (float*)d_out;                // [2048,16,128]
    (void)in_sizes; (void)n_in; (void)out_size; (void)d_ws; (void)ws_size;

    interaction_mlp_kernel<<<dim3(2048), dim3(256), 0, stream>>>(
        x, W1, b1, W2, b2, out);
}
